// ConvBlock_20633022890307
// MI455X (gfx1250) — matrix operand order repaired and checked by reading, not compiled, not run on hardware
//
#include <hip/hip_runtime.h>
#include <hip/hip_bf16.h>

#define N_VOX 400000
#define C_IN 64
#define C_OUT 64
#define KVOL 27
#define EPS_BN 1e-5f
#define GRID1 512
#define BLOCK1 256   // 8 waves of 32
#define WPB (BLOCK1 / 32)

typedef _Float16 f16;
typedef f16   v4h  __attribute__((ext_vector_type(4)));
typedef f16   v8h  __attribute__((ext_vector_type(8)));
typedef f16   v16h __attribute__((ext_vector_type(16)));
typedef float v8f  __attribute__((ext_vector_type(8)));

static __device__ __forceinline__ v16h cat16(v8h lo, v8h hi) {
    return __builtin_shufflevector(lo, hi, 0,1,2,3,4,5,6,7,8,9,10,11,12,13,14,15);
}

// A fragment gather: 16-bit A 16x32 layout -> lane holds channels
// {kc*32 + half*8 + j} (e0..7) and {kc*32 + 16 + half*8 + j} (e8..15)
static __device__ __forceinline__ void load_a(const f16* __restrict__ feat, int row,
                                              int half, v16h& a_kc0, v16h& a_kc1) {
    const v8h* p = (const v8h*)(feat + (size_t)row * C_IN);
    a_kc0 = cat16(p[half],     p[half + 2]);
    a_kc1 = cat16(p[half + 4], p[half + 6]);
}

// ---------- 1) features f32 -> f16, plus one zeroed dummy row at index N_VOX ----------
__global__ void cvt_feat(const float* __restrict__ f, f16* __restrict__ o) {
    const long n4 = (long)N_VOX * (C_IN / 4);
    const long t4 = (long)(N_VOX + 1) * (C_IN / 4);
    for (long i = (long)blockIdx.x * blockDim.x + threadIdx.x; i < t4;
         i += (long)gridDim.x * blockDim.x) {
        float4 v = make_float4(0.f, 0.f, 0.f, 0.f);
        if (i < n4) v = ((const float4*)f)[i];
        v4h h = { (f16)v.x, (f16)v.y, (f16)v.z, (f16)v.w };
        ((v4h*)o)[i] = h;
    }
}

// ---------- 2) swizzle weight[27][64][64] (k,K,n) into WMMA-B lane layout ----------
// f16 flat index: ((((k*2+kc)*4+nt)*2+c)*32 + lane)*8 + e8
//   value = W[k][kc*32 + c*16 + (lane>>4)*8 + e8][nt*16 + (lane&15)]
// => per-lane B fragment = two 16B chunks at lane-stride 16B (LDS bank friendly)
__global__ void swz_w(const float* __restrict__ w, f16* __restrict__ o) {
    const int total = KVOL * 4096;
    for (int i = blockIdx.x * blockDim.x + threadIdx.x; i < total;
         i += gridDim.x * blockDim.x) {
        int e8   = i & 7;
        int lane = (i >> 3) & 31;
        int c    = (i >> 8) & 1;
        int nt   = (i >> 9) & 3;
        int kc   = (i >> 11) & 1;
        int k    = i >> 12;
        int n = nt * 16 + (lane & 15);
        int K = kc * 32 + c * 16 + (lane >> 4) * 8 + e8;
        o[i] = (f16)w[(k * C_IN + K) * C_OUT + n];
    }
}

// ---------- 3) gathered GEMM: LDS-staged double-buffered B + pipelined A gather ----------
__global__ void __launch_bounds__(BLOCK1) conv_wmma(
        const int* __restrict__ nmap, const f16* __restrict__ feat,
        const f16* __restrict__ wsw, float* __restrict__ out,
        float* __restrict__ partial) {
    __shared__ f16 wlds[2][4096];            // 2 x 8KB weight slabs
    __shared__ float lds_sum[WPB][C_OUT];
    __shared__ float lds_ssq[WPB][C_OUT];

    const int tid  = threadIdx.x;
    const int lane = tid & 31;
    const int wv   = tid >> 5;
    const int lm   = lane & 15;   // N-column / voxel-row within half
    const int half = lane >> 4;

    const int njobs = (N_VOX + 31) / 32;
    const int njb   = (njobs + WPB - 1) / WPB;   // block-level job tiles (uniform barriers)
    const uint4* gsw = (const uint4*)wsw;         // 512 x 16B units per k

    float csum[4] = {0.f, 0.f, 0.f, 0.f};
    float cssq[4] = {0.f, 0.f, 0.f, 0.f};

    for (int jb = blockIdx.x; jb < njb; jb += gridDim.x) {
        const int job  = jb * WPB + wv;          // may be >= njobs: guarded below
        const int base = job * 32;

        v8f acc[2][4];
#pragma unroll
        for (int mt = 0; mt < 2; ++mt)
#pragma unroll
            for (int nt = 0; nt < 4; ++nt) { v8f z = {}; acc[mt][nt] = z; }

        const int r0 = base + lm;
        const int r1 = base + 16 + lm;
        const bool ok0 = (r0 < N_VOX), ok1 = (r1 < N_VOX);
        const int rr0 = ok0 ? r0 : 0;
        const int rr1 = ok1 ? r1 : 0;

        // ---- stage k=0 weights into buffer 0, preload k=0 A fragments ----
        {
            uint4 w0 = gsw[tid], w1 = gsw[tid + 256];
            ((uint4*)wlds[0])[tid]       = w0;
            ((uint4*)wlds[0])[tid + 256] = w1;
        }
        int i0 = nmap[rr0];
        int i1 = nmap[rr1];
        int row0 = (i0 < 0 || !ok0) ? N_VOX : i0;
        int row1 = (i1 < 0 || !ok1) ? N_VOX : i1;
        v16h a[2][2], an[2][2];                   // [mt][kc]
        load_a(feat, row0, half, a[0][0], a[0][1]);
        load_a(feat, row1, half, a[1][0], a[1][1]);
        __syncthreads();

        for (int k = 0; k < KVOL; ++k) {
            // ---- prefetch k+1 (clamped; wasted only on last iter) ----
            const int kn = (k + 1 < KVOL) ? k + 1 : KVOL - 1;
            uint4 wn0 = gsw[kn * 512 + tid];
            uint4 wn1 = gsw[kn * 512 + tid + 256];
            int ni0 = nmap[kn * N_VOX + rr0];
            int ni1 = nmap[kn * N_VOX + rr1];
            int nrow0 = (ni0 < 0 || !ok0) ? N_VOX : ni0;
            int nrow1 = (ni1 < 0 || !ok1) ? N_VOX : ni1;
            load_a(feat, nrow0, half, an[0][0], an[0][1]);
            load_a(feat, nrow1, half, an[1][0], an[1][1]);

            // ---- compute with current slab: B from LDS, 16 WMMAs ----
            const v8h* wb = (const v8h*)wlds[k & 1];
#pragma unroll
            for (int kc = 0; kc < 2; ++kc) {
#pragma unroll
                for (int nt = 0; nt < 4; ++nt) {
                    const int cbase = ((kc * 4 + nt) * 2) * 32 + lane;
                    v16h b = cat16(wb[cbase], wb[cbase + 32]);
                    acc[0][nt] = __builtin_amdgcn_wmma_f32_16x16x32_f16(
                        false, a[0][kc], false, b, (short)0, acc[0][nt], false, false);
                    acc[1][nt] = __builtin_amdgcn_wmma_f32_16x16x32_f16(
                        false, a[1][kc], false, b, (short)0, acc[1][nt], false, false);
                }
            }

            // ---- store prefetched slab to the other buffer; one barrier per k ----
            ((uint4*)wlds[(k + 1) & 1])[tid]       = wn0;
            ((uint4*)wlds[(k + 1) & 1])[tid + 256] = wn1;
            __syncthreads();

#pragma unroll
            for (int mt = 0; mt < 2; ++mt)
#pragma unroll
                for (int kc = 0; kc < 2; ++kc) a[mt][kc] = an[mt][kc];
        }

        // ---- epilogue: D lane->(N=lm), vgpr j->(M = half*8 + j) ----
#pragma unroll
        for (int mt = 0; mt < 2; ++mt) {
            const int rbase = base + mt * 16 + half * 8;
#pragma unroll
            for (int nt = 0; nt < 4; ++nt) {
                const int col = nt * 16 + lm;
#pragma unroll
                for (int j = 0; j < 8; ++j) {
                    const int row = rbase + j;
                    if (row < N_VOX) {
                        float v = acc[mt][nt][j];
                        out[(size_t)row * C_OUT + col] = v;
                        csum[nt] += v;
                        cssq[nt] += v * v;
                    }
                }
            }
        }
    }

    // ---- deterministic block stats: lanes l and l+16 share channel nt*16+lm ----
#pragma unroll
    for (int nt = 0; nt < 4; ++nt) {
        float s = csum[nt] + __shfl_xor(csum[nt], 16, 32);
        float q = cssq[nt] + __shfl_xor(cssq[nt], 16, 32);
        if (half == 0) {
            lds_sum[wv][nt * 16 + lm] = s;
            lds_ssq[wv][nt * 16 + lm] = q;
        }
    }
    __syncthreads();
    if (tid < C_OUT) {
        float s = 0.f, q = 0.f;
#pragma unroll
        for (int w8 = 0; w8 < WPB; ++w8) {
            s += lds_sum[w8][tid];
            q += lds_ssq[w8][tid];
        }
        partial[blockIdx.x * 128 + tid]      = s;
        partial[blockIdx.x * 128 + 64 + tid] = q;
    }
}

// ---------- 4) reduce partials -> fused scale/shift ----------
__global__ void stats_k(const float* __restrict__ partial,
                        const float* __restrict__ gamma,
                        const float* __restrict__ beta,
                        float* __restrict__ scsh) {
    int c = threadIdx.x;
    if (c >= C_OUT) return;
    float s = 0.f, q = 0.f;
    for (int b = 0; b < GRID1; ++b) {
        s += partial[b * 128 + c];
        q += partial[b * 128 + 64 + c];
    }
    const float inv  = 1.0f / (float)N_VOX;
    float mean = s * inv;
    float var  = fmaxf(q * inv - mean * mean, 0.f);
    float rstd = rsqrtf(var + EPS_BN);
    float sc   = gamma[c] * rstd;
    scsh[c]      = sc;
    scsh[64 + c] = beta[c] - mean * sc;
}

// ---------- 5) in-place BN + ReLU ----------
__global__ void bnrelu_k(float* __restrict__ out, const float* __restrict__ scsh) {
    const long t4 = (long)N_VOX * C_OUT / 4;
    for (long i = (long)blockIdx.x * blockDim.x + threadIdx.x; i < t4;
         i += (long)gridDim.x * blockDim.x) {
        float4 v = ((float4*)out)[i];
        int c = (int)((i * 4) & (C_OUT - 1));
        v.x = fmaxf(v.x * scsh[c + 0] + scsh[64 + c + 0], 0.f);
        v.y = fmaxf(v.y * scsh[c + 1] + scsh[64 + c + 1], 0.f);
        v.z = fmaxf(v.z * scsh[c + 2] + scsh[64 + c + 2], 0.f);
        v.w = fmaxf(v.w * scsh[c + 3] + scsh[64 + c + 3], 0.f);
        ((float4*)out)[i] = v;
    }
}

extern "C" void kernel_launch(void* const* d_in, const int* in_sizes, int n_in,
                              void* d_out, int out_size, void* d_ws, size_t ws_size,
                              hipStream_t stream) {
    (void)in_sizes; (void)n_in; (void)out_size; (void)ws_size;
    const float* features = (const float*)d_in[0];
    const int*   nmap     = (const int*)d_in[1];
    const float* weight   = (const float*)d_in[2];
    const float* gamma    = (const float*)d_in[3];
    const float* beta     = (const float*)d_in[4];
    float* out = (float*)d_out;

    char* ws = (char*)d_ws;
    size_t off = 0;
    f16* feat16 = (f16*)(ws + off);
    off += (size_t)(N_VOX + 1) * C_IN * sizeof(f16);
    off = (off + 255) & ~(size_t)255;
    f16* wsw = (f16*)(ws + off);
    off += (size_t)KVOL * 4096 * sizeof(f16);
    off = (off + 255) & ~(size_t)255;
    float* partial = (float*)(ws + off);
    off += (size_t)GRID1 * 128 * sizeof(float);
    float* scsh = (float*)(ws + off);

    hipLaunchKernelGGL(cvt_feat, dim3(2048), dim3(256), 0, stream, features, feat16);
    hipLaunchKernelGGL(swz_w, dim3(432), dim3(256), 0, stream, weight, wsw);
    hipLaunchKernelGGL(conv_wmma, dim3(GRID1), dim3(BLOCK1), 0, stream,
                       nmap, feat16, wsw, out, partial);
    hipLaunchKernelGGL(stats_k, dim3(1), dim3(64), 0, stream, partial, gamma, beta, scsh);
    hipLaunchKernelGGL(bnrelu_k, dim3(2048), dim3(256), 0, stream, out, scsh);
}
